// Discriminator_43911745635124
// MI455X (gfx1250) — matrix operand order repaired and checked by reading, not compiled, not run on hardware
//
#include <hip/hip_runtime.h>
#include <hip/hip_fp16.h>
#include <math.h>

typedef __attribute__((ext_vector_type(16))) _Float16 v16h;
typedef __attribute__((ext_vector_type(8)))  _Float16 v8h;
typedef __attribute__((ext_vector_type(8)))  float    v8f;
typedef int v4i_ __attribute__((vector_size(16)));

#define B_  64
#define S_  256
#define V_  16
#define D_  512
#define F_  2048
#define L_  6
#define DL_ 256
#define H_  8
#define DH_ 64

#if defined(__has_builtin)
#if __has_builtin(__builtin_amdgcn_global_load_async_to_lds_b128) && \
    __has_builtin(__builtin_amdgcn_s_wait_asynccnt)
#define USE_ASYNC_LDS 1
#endif
#endif
#ifndef USE_ASYNC_LDS
#define USE_ASYNC_LDS 0
#endif

#if USE_ASYNC_LDS
typedef __attribute__((address_space(1))) v4i_* gv4p;   // global v4i*
typedef __attribute__((address_space(3))) v4i_* lv4p;   // LDS v4i*
#endif

union FragH { v16h v; v8h h[2]; uint4 u[2]; };

// ---------------- generic f32 -> f16 convert ----------------
__global__ void cvt_f32_to_f16(const float* __restrict__ in, _Float16* __restrict__ out, long n) {
  long i = (long)blockIdx.x * blockDim.x + threadIdx.x;
  long stride = (long)gridDim.x * blockDim.x;
  for (; i < n; i += stride) out[i] = (_Float16)in[i];
}

// ---------------- embedding + positional encoding ----------------
__global__ void embed_kernel(const float* __restrict__ src, const float* __restrict__ ew,
                             const float* __restrict__ eb, float* __restrict__ x,
                             _Float16* __restrict__ xh) {
  long idx = (long)blockIdx.x * blockDim.x + threadIdx.x;  // over B*S*D
  if (idx >= (long)B_ * S_ * D_) return;
  int  d  = (int)(idx & (D_ - 1));
  long bs = idx >> 9;                 // D_=512
  int  s  = (int)(bs & (S_ - 1));
  float acc = eb[d];
  const float* sp = src + bs * V_;
  const float* wp = ew + (long)d * V_;
#pragma unroll
  for (int v = 0; v < V_; ++v) acc += sp[v] * wp[v];
  const float c = 9.210340371976184f / (float)D_;   // ln(10000)/D
  float den = __expf(-(float)(d & ~1) * c);
  float ang = (float)s * den;
  acc += (d & 1) ? __cosf(ang) : __sinf(ang);
  x[idx]  = acc;
  xh[idx] = (_Float16)acc;
}

// ---------------- WMMA GEMM:  C = alpha * A[M,K] * W[N,K]^T + bias ----------------
// A, W row-major f16.  mode: 0 = f16 out, 1 = f16 out + relu, 2 = f32 out.
// Batched via blockIdx.z: offset = (z/zdiv)*O + (z%zdiv)*I   (elements).
// Tile: 128x128, BK=64, 8 waves (2x4), wave tile 64x32, double-buffered LDS,
// staged with GLOBAL_LOAD_ASYNC_TO_LDS_B128 when available (ASYNCcnt).
#define BK_   64
#define LDSP_ 72   // row pitch in halves (144B: 16B-aligned, conflict-free frags)

__global__ __launch_bounds__(256)
void gemm_wmma(const _Float16* __restrict__ A, long lda, long aO, long aI,
               const _Float16* __restrict__ Bw, long ldb, long bO, long bI,
               const float* __restrict__ bias,
               _Float16* __restrict__ C16, float* __restrict__ C32,
               long ldc, long cO, long cI,
               int M, int N, int K, int zdiv, float alpha, int mode) {
  __shared__ _Float16 sA[2][128 * LDSP_];
  __shared__ _Float16 sB[2][128 * LDSP_];

  int zb = blockIdx.z / zdiv, zi = blockIdx.z % zdiv;
  A  += (long)zb * aO + (long)zi * aI;
  Bw += (long)zb * bO + (long)zi * bI;
  long coff = (long)zb * cO + (long)zi * cI;

  int bm = blockIdx.y * 128;
  int bn = blockIdx.x * 128;
  int t = threadIdx.x;
  int lane = t & 31, wid = t >> 5;
  int wm = (wid & 1) * 64;     // wave tile: 64 rows (M)
  int wn = (wid >> 1) * 32;    // 32 cols (N)
  int lr = lane & 15, hi = lane >> 4;

  int lrow = t >> 1, lch = t & 1;   // loader: row 0..127, 64B chunk 0..1

  // Pre-zero guard rows (async loads are EXEC-masked; skipped rows must be 0).
  {
    uint4 z4 = {0, 0, 0, 0};
    if (bm + lrow >= M) {
#pragma unroll
      for (int bb = 0; bb < 2; ++bb) {
        uint4* dz = (uint4*)(&sA[bb][lrow * LDSP_ + lch * 32]);
        dz[0] = z4; dz[1] = z4; dz[2] = z4; dz[3] = z4;
      }
    }
    if (bn + lrow >= N) {
#pragma unroll
      for (int bb = 0; bb < 2; ++bb) {
        uint4* dz = (uint4*)(&sB[bb][lrow * LDSP_ + lch * 32]);
        dz[0] = z4; dz[1] = z4; dz[2] = z4; dz[3] = z4;
      }
    }
  }

  v8f acc[4][2];
#pragma unroll
  for (int i = 0; i < 4; ++i)
#pragma unroll
    for (int j = 0; j < 2; ++j)
#pragma unroll
      for (int q = 0; q < 8; ++q) acc[i][j][q] = 0.0f;

  auto stage = [&](int buf, int kt) {
    long grA = bm + lrow;
    long grB = bn + lrow;
#if USE_ASYNC_LDS
    if (grA < M) {
      const _Float16* src = A + grA * lda + kt + lch * 32;
      _Float16* dst = &sA[buf][lrow * LDSP_ + lch * 32];
#pragma unroll
      for (int j = 0; j < 4; ++j)
        __builtin_amdgcn_global_load_async_to_lds_b128(
            (gv4p)(src + j * 8), (lv4p)(dst + j * 8), 0, 0);
    }
    if (grB < N) {
      const _Float16* src = Bw + grB * ldb + kt + lch * 32;
      _Float16* dst = &sB[buf][lrow * LDSP_ + lch * 32];
#pragma unroll
      for (int j = 0; j < 4; ++j)
        __builtin_amdgcn_global_load_async_to_lds_b128(
            (gv4p)(src + j * 8), (lv4p)(dst + j * 8), 0, 0);
    }
#else
    uint4 z4 = {0, 0, 0, 0};
    {
      uint4 d0 = z4, d1 = z4, d2 = z4, d3 = z4;
      if (grA < M) {
        const uint4* p = (const uint4*)(A + grA * lda + kt + lch * 32);
        d0 = p[0]; d1 = p[1]; d2 = p[2]; d3 = p[3];
      }
      uint4* dst = (uint4*)(&sA[buf][lrow * LDSP_ + lch * 32]);
      dst[0] = d0; dst[1] = d1; dst[2] = d2; dst[3] = d3;
    }
    {
      uint4 d0 = z4, d1 = z4, d2 = z4, d3 = z4;
      if (grB < N) {
        const uint4* p = (const uint4*)(Bw + grB * ldb + kt + lch * 32);
        d0 = p[0]; d1 = p[1]; d2 = p[2]; d3 = p[3];
      }
      uint4* dst = (uint4*)(&sB[buf][lrow * LDSP_ + lch * 32]);
      dst[0] = d0; dst[1] = d1; dst[2] = d2; dst[3] = d3;
    }
#endif
  };

  auto tile_sync = [&]() {
#if USE_ASYNC_LDS
    __builtin_amdgcn_s_wait_asynccnt(0);
#endif
    __syncthreads();
  };

  auto compute = [&](int buf) {
    const _Float16* As = sA[buf];
    const _Float16* Bs = sB[buf];
#pragma unroll
    for (int kk = 0; kk < BK_; kk += 32) {
      FragH afr[4], bfr[2];
#pragma unroll
      for (int ms = 0; ms < 4; ++ms) {
        // A 16x32 frag: lanes 0-15 K={0..7,16..23}, lanes 16-31 K={8..15,24..31}
        const _Float16* p = As + (wm + ms * 16 + lr) * LDSP_ + kk + hi * 8;
        afr[ms].h[0] = *(const v8h*)(p);
        afr[ms].h[1] = *(const v8h*)(p + 16);
      }
#pragma unroll
      for (int ns = 0; ns < 2; ++ns) {
        // B 32x16 frag: lanes 0-15 K={0..7,16..23}, lanes 16-31 K={8..15,24..31}
        const _Float16* p = Bs + (wn + ns * 16 + lr) * LDSP_ + kk + hi * 8;
        bfr[ns].h[0] = *(const v8h*)(p);
        bfr[ns].h[1] = *(const v8h*)(p + 16);
      }
#pragma unroll
      for (int ms = 0; ms < 4; ++ms)
#pragma unroll
        for (int ns = 0; ns < 2; ++ns)
          acc[ms][ns] = __builtin_amdgcn_wmma_f32_16x16x32_f16(
              false, afr[ms].v, false, bfr[ns].v, (short)0, acc[ms][ns], false, false);
    }
  };

  int nk = K >> 6;   // K is a multiple of 64 for every call site
  stage(0, 0);
  tile_sync();
  int cur = 0;
  for (int kt = 1; kt < nk; ++kt) {
    stage(cur ^ 1, kt << 6);
    compute(cur);
    tile_sync();
    cur ^= 1;
  }
  compute(cur);

  // epilogue: C frag: col = lane&15, row = q + 8*(lane>>4)
#pragma unroll
  for (int ms = 0; ms < 4; ++ms) {
#pragma unroll
    for (int ns = 0; ns < 2; ++ns) {
      int gcol = bn + wn + ns * 16 + lr;
      if (gcol >= N) continue;
      float bv = bias ? bias[gcol] : 0.0f;
#pragma unroll
      for (int q = 0; q < 8; ++q) {
        int grow = bm + wm + ms * 16 + hi * 8 + q;
        if (grow >= M) continue;
        float v = acc[ms][ns][q] * alpha + bv;
        if (mode == 1 && v < 0.0f) v = 0.0f;
        long off = coff + (long)grow * ldc + gcol;
        if (mode == 2) C32[off] = v;
        else           C16[off] = (_Float16)v;
      }
    }
  }
}

// ---------------- softmax over 256 cols, in place on f16 ----------------
__global__ __launch_bounds__(256) void softmax_kernel(_Float16* __restrict__ s) {
  __shared__ float red[256];
  long row = blockIdx.x;
  int t = threadIdx.x;
  _Float16* p = s + row * 256;
  float v = (float)p[t];
  red[t] = v; __syncthreads();
  for (int off = 128; off > 0; off >>= 1) { if (t < off) red[t] = fmaxf(red[t], red[t + off]); __syncthreads(); }
  float mx = red[0]; __syncthreads();
  float e = __expf(v - mx);
  red[t] = e; __syncthreads();
  for (int off = 128; off > 0; off >>= 1) { if (t < off) red[t] += red[t + off]; __syncthreads(); }
  float inv = 1.0f / red[0];
  p[t] = (_Float16)(e * inv);
}

// ---------------- fused residual + LayerNorm (D=512, 256 thr x 2) ----------------
__global__ __launch_bounds__(256)
void ln_kernel(float* __restrict__ x, const float* __restrict__ delta,
               const float* __restrict__ g, const float* __restrict__ b,
               _Float16* __restrict__ xh) {
  __shared__ float red[256];
  long row = blockIdx.x;
  int t = threadIdx.x;
  long base = row * D_;
  float v0 = x[base + t]       + delta[base + t];
  float v1 = x[base + t + 256] + delta[base + t + 256];
  red[t] = v0 + v1; __syncthreads();
  for (int off = 128; off > 0; off >>= 1) { if (t < off) red[t] += red[t + off]; __syncthreads(); }
  float mu = red[0] * (1.0f / D_); __syncthreads();
  float d0 = v0 - mu, d1 = v1 - mu;
  red[t] = d0 * d0 + d1 * d1; __syncthreads();
  for (int off = 128; off > 0; off >>= 1) { if (t < off) red[t] += red[t + off]; __syncthreads(); }
  float rstd = rsqrtf(red[0] * (1.0f / D_) + 1e-5f);
  float y0 = d0 * rstd * g[t] + b[t];
  float y1 = d1 * rstd * g[t + 256] + b[t + 256];
  x[base + t] = y0;              x[base + t + 256] = y1;
  xh[base + t] = (_Float16)y0;   xh[base + t + 256] = (_Float16)y1;
}

// ---------------- v transpose: qkv[b,s, 2D + h*dh + d] -> vT[b,h,d,s] ----------------
__global__ void transpose_v(const _Float16* __restrict__ qkv, _Float16* __restrict__ vT) {
  long idx = (long)blockIdx.x * blockDim.x + threadIdx.x;   // over B*H*DH*S
  if (idx >= (long)B_ * H_ * DH_ * S_) return;
  int  s  = (int)(idx & (S_ - 1));
  long r  = idx >> 8;
  int  d  = (int)(r & (DH_ - 1));
  long bh = r >> 6;
  int  h  = (int)(bh & (H_ - 1));
  long b  = bh >> 3;
  vT[idx] = qkv[b * (long)S_ * 3 * D_ + (long)s * 3 * D_ + 2 * D_ + h * DH_ + d];
}

// ---------------- latent: mem[b,j] = sum_{i<131072} xh[b,i]*latw[j,i] + lb[j] ----------------
__global__ __launch_bounds__(256)
void latent_kernel(const _Float16* __restrict__ xh, const _Float16* __restrict__ lw,
                   const float* __restrict__ lb, float* __restrict__ mem) {
  int wid  = blockIdx.x * 8 + (threadIdx.x >> 5);   // global wave id over B*DL = 16384
  int lane = threadIdx.x & 31;
  int b = wid >> 8;
  int j = wid & (DL_ - 1);
  const _Float16* xp = xh + (long)b * (S_ * D_);
  const _Float16* wp = lw + (long)j * (S_ * D_);
  float acc = 0.0f;
  for (long i = (long)lane * 8; i < (long)S_ * D_; i += 32 * 8) {
    v8h a = *(const v8h*)(xp + i);
    v8h w = *(const v8h*)(wp + i);
#pragma unroll
    for (int e = 0; e < 8; ++e) acc += (float)a[e] * (float)w[e];
  }
  for (int off = 16; off > 0; off >>= 1) acc += __shfl_down(acc, off, 32);
  if (lane == 0) mem[wid] = acc + lb[j];
}

// ---------------- head: sigmoid(mem @ head_w^T + head_b) ----------------
__global__ void head_kernel(const float* __restrict__ mem, const float* __restrict__ hw,
                            const float* __restrict__ hb, float* __restrict__ out) {
  int b = threadIdx.x;
  if (b >= B_) return;
  float acc = hb[0];
  for (int i = 0; i < DL_; ++i) acc += mem[b * DL_ + i] * hw[i];
  out[b] = 1.0f / (1.0f + __expf(-acc));
}

// ---------------- host side ----------------
static void launch_gemm(hipStream_t st,
                        const _Float16* A, long lda, long aO, long aI,
                        const _Float16* Bw, long ldb, long bO, long bI,
                        const float* bias,
                        _Float16* C16, float* C32, long ldc, long cO, long cI,
                        int M, int N, int K, int Z, int zdiv, float alpha, int mode) {
  dim3 g((N + 127) / 128, (M + 127) / 128, Z);
  gemm_wmma<<<g, 256, 0, st>>>(A, lda, aO, aI, Bw, ldb, bO, bI, bias,
                               C16, C32, ldc, cO, cI, M, N, K, zdiv, alpha, mode);
}

extern "C" void kernel_launch(void* const* d_in, const int* in_sizes, int n_in,
                              void* d_out, int out_size, void* d_ws, size_t ws_size,
                              hipStream_t stream) {
  (void)in_sizes; (void)n_in; (void)out_size; (void)ws_size;
  const float* src    = (const float*)d_in[0];
  const float* emb_w  = (const float*)d_in[1];
  const float* emb_b  = (const float*)d_in[2];
  const float* qkv_w  = (const float*)d_in[3];
  const float* qkv_b  = (const float*)d_in[4];
  const float* out_w  = (const float*)d_in[5];
  const float* out_b  = (const float*)d_in[6];
  const float* ff1_w  = (const float*)d_in[7];
  const float* ff1_b  = (const float*)d_in[8];
  const float* ff2_w  = (const float*)d_in[9];
  const float* ff2_b  = (const float*)d_in[10];
  const float* ln1_g  = (const float*)d_in[11];
  const float* ln1_b  = (const float*)d_in[12];
  const float* ln2_g  = (const float*)d_in[13];
  const float* ln2_b  = (const float*)d_in[14];
  const float* lat_w  = (const float*)d_in[15];
  const float* lat_b  = (const float*)d_in[16];
  const float* head_w = (const float*)d_in[17];
  const float* head_b = (const float*)d_in[18];
  float* pred = (float*)d_out;

  char* ws = (char*)d_ws;
  size_t off = 0;
  auto alloc = [&](size_t bytes) -> void* {
    void* p = ws + off;
    off = (off + bytes + 255) & ~(size_t)255;
    return p;
  };
  const long MD = (long)B_ * S_ * D_;          // 8,388,608
  _Float16* qkvw_h = (_Float16*)alloc((size_t)L_ * 3 * D_ * D_ * 2);
  _Float16* outw_h = (_Float16*)alloc((size_t)L_ * D_ * D_ * 2);
  _Float16* ff1w_h = (_Float16*)alloc((size_t)L_ * F_ * D_ * 2);
  _Float16* ff2w_h = (_Float16*)alloc((size_t)L_ * D_ * F_ * 2);
  float*    x      = (float*)   alloc((size_t)MD * 4);
  _Float16* xh     = (_Float16*)alloc((size_t)MD * 2);
  _Float16* qkvh   = (_Float16*)alloc((size_t)B_ * S_ * 3 * D_ * 2);
  _Float16* vT     = (_Float16*)alloc((size_t)B_ * H_ * DH_ * S_ * 2);
  _Float16* oh     = (_Float16*)alloc((size_t)MD * 2);
  float*    delta  = (float*)   alloc((size_t)MD * 4);
  // U is reused: attention scores (B*H*S*S f16) -> FFN hidden (B*S*F f16) -> lat_w f16
  _Float16* U      = (_Float16*)alloc((size_t)B_ * H_ * S_ * S_ * 2);
  float*    memb   = (float*)   alloc((size_t)B_ * DL_ * 4);

  // --- weight conversion to f16 ---
  cvt_f32_to_f16<<<4096, 256, 0, stream>>>(qkv_w, qkvw_h, (long)L_ * 3 * D_ * D_);
  cvt_f32_to_f16<<<4096, 256, 0, stream>>>(out_w, outw_h, (long)L_ * D_ * D_);
  cvt_f32_to_f16<<<4096, 256, 0, stream>>>(ff1_w, ff1w_h, (long)L_ * F_ * D_);
  cvt_f32_to_f16<<<4096, 256, 0, stream>>>(ff2_w, ff2w_h, (long)L_ * D_ * F_);

  // --- embedding + posenc ---
  embed_kernel<<<(int)(MD / 256), 256, 0, stream>>>(src, emb_w, emb_b, x, xh);

  const int M = B_ * S_;      // 16384
  for (int l = 0; l < L_; ++l) {
    // qkv = xh @ qkv_w[l]^T + qkv_b[l]   -> f16 [B,S,3D]
    launch_gemm(stream, xh, D_, 0, 0,
                qkvw_h + (size_t)l * 3 * D_ * D_, D_, 0, 0,
                qkv_b + (size_t)l * 3 * D_,
                qkvh, nullptr, 3 * D_, 0, 0,
                M, 3 * D_, D_, 1, 1, 1.0f, 0);

    // v transpose for AV gemm
    transpose_v<<<(int)(((long)B_ * H_ * DH_ * S_) / 256), 256, 0, stream>>>(qkvh, vT);

    // scores[b,h] = (q @ k^T) * 1/sqrt(dh)   (z = b*H + h)
    launch_gemm(stream, qkvh, 3 * D_, (long)S_ * 3 * D_, DH_,
                qkvh + D_, 3 * D_, (long)S_ * 3 * D_, DH_,
                nullptr,
                U, nullptr, S_, (long)H_ * S_ * S_, (long)S_ * S_,
                S_, S_, DH_, B_ * H_, H_, 0.125f, 0);

    softmax_kernel<<<B_ * H_ * S_, 256, 0, stream>>>(U);

    // o[b,s,h*dh+d] = P @ v      (z = b*H + h)
    launch_gemm(stream, U, S_, (long)H_ * S_ * S_, (long)S_ * S_,
                vT, S_, (long)H_ * DH_ * S_, (long)DH_ * S_,
                nullptr,
                oh, nullptr, D_, (long)S_ * D_, DH_,
                S_, DH_, S_, B_ * H_, H_, 1.0f, 0);

    // delta = o @ out_w[l]^T + out_b[l]   -> f32
    launch_gemm(stream, oh, D_, 0, 0,
                outw_h + (size_t)l * D_ * D_, D_, 0, 0,
                out_b + (size_t)l * D_,
                nullptr, delta, D_, 0, 0,
                M, D_, D_, 1, 1, 1.0f, 2);

    ln_kernel<<<M, 256, 0, stream>>>(x, delta, ln1_g + (size_t)l * D_, ln1_b + (size_t)l * D_, xh);

    // h = relu(xh @ ff1_w[l]^T + ff1_b[l])  -> f16 (reuses U)
    launch_gemm(stream, xh, D_, 0, 0,
                ff1w_h + (size_t)l * F_ * D_, D_, 0, 0,
                ff1_b + (size_t)l * F_,
                U, nullptr, F_, 0, 0,
                M, F_, D_, 1, 1, 1.0f, 1);

    // delta = h @ ff2_w[l]^T + ff2_b[l]  -> f32
    launch_gemm(stream, U, F_, 0, 0,
                ff2w_h + (size_t)l * D_ * F_, F_, 0, 0,
                ff2_b + (size_t)l * D_,
                nullptr, delta, D_, 0, 0,
                M, D_, F_, 1, 1, 1.0f, 2);

    ln_kernel<<<M, 256, 0, stream>>>(x, delta, ln2_g + (size_t)l * D_, ln2_b + (size_t)l * D_, xh);
  }

  // --- latent projection (convert lat_w now; U is free) ---
  cvt_f32_to_f16<<<8192, 256, 0, stream>>>(lat_w, U, (long)DL_ * S_ * D_);
  latent_kernel<<<(B_ * DL_) / 8, 256, 0, stream>>>(xh, U, lat_b, memb);

  head_kernel<<<1, 64, 0, stream>>>(memb, head_w, head_b, pred);
}
